// MBartAttention_71571335021070
// MI455X (gfx1250) — matrix operand order repaired and checked by reading, not compiled, not run on hardware
//
#include <hip/hip_runtime.h>
#include <hip/hip_bf16.h>

// ---------------------------------------------------------------------------
// MBart attention forward on MI455X (gfx1250), bf16 WMMA + f32 accumulate.
// T=2048 B=2 E=1024 H=16 HD=64, low-rank (R=4) weight modulation.
// GEMM mainloop is software-pipelined with ping-pong fragment buffers
// (manual 2x unroll) so no register copies are needed for the rotation and
// loads of step k+1 overlap the WMMAs of step k.
// ---------------------------------------------------------------------------

typedef __attribute__((ext_vector_type(16))) __bf16 v16bf;
typedef __attribute__((ext_vector_type(8)))  __bf16 v8bf;
typedef __attribute__((ext_vector_type(8)))  float  v8f;

#define T_DIM   2048
#define B_DIM   2
#define E_DIM   1024
#define H_DIM   16
#define HD_DIM  64
#define E3_DIM  3072
#define M_DIM   (T_DIM * B_DIM)   // 4096 flattened (t,b) rows
#define SCALE_Q 0.125f            // HD^-0.5
#define NEG_BIG (-1.0e9f)

static __device__ __forceinline__ v16bf cat8(v8bf a, v8bf b) {
  return __builtin_shufflevector(a, b, 0,1,2,3,4,5,6,7,8,9,10,11,12,13,14,15);
}

// A fragment (16x32 bf16). Per ISA: lane%16 = M row; per-lane K order is
// two contiguous 8-chunks: K = 8*(lane/16)+0..7 and K = 16+8*(lane/16)+0..7.
static __device__ __forceinline__ v16bf load_a_frag(const __bf16* __restrict__ A,
                                                    int ld, int m0, int k0, int lane) {
  const __bf16* base = A + (long)(m0 + (lane & 15)) * ld + k0 + ((lane >> 4) << 3);
  v8bf lo = *(const v8bf*)(base);
  v8bf hi = *(const v8bf*)(base + 16);
  return cat8(lo, hi);
}

// B fragment (32x16 bf16) with B[k][n] = W[n0+n][k0+k] (W row-major, ld elems).
// Per ISA: lane%16 = N column; per-lane K order matches A: K = 8*(lane/16)+0..7 and K = 16+8*(lane/16)+0..7.
static __device__ __forceinline__ v16bf load_b_frag(const __bf16* __restrict__ W,
                                                    int ld, int n0, int k0, int lane) {
  const __bf16* base = W + (long)(n0 + (lane & 15)) * ld + k0 + ((lane >> 4) << 3);
  v8bf lo = *(const v8bf*)(base);
  v8bf hi = *(const v8bf*)(base + 16);
  return cat8(lo, hi);
}

// ---------------------------------------------------------------------------
// Kernel 1: modulated weights  w = pw * (rm^T sm) + (r^T s)  -> bf16
// ---------------------------------------------------------------------------
__global__ __launch_bounds__(256)
void build_w_kernel(const float* __restrict__ pw,
                    const float* __restrict__ rm, const float* __restrict__ sm,
                    const float* __restrict__ ra, const float* __restrict__ sa,
                    const int* __restrict__ lang,
                    __bf16* __restrict__ w, int rows) {
  int idx = blockIdx.x * blockDim.x + threadIdx.x;   // over rows*E
  int e = idx / E_DIM, f = idx % E_DIM;
  int li = lang[0];
  const float* rmL = rm + (long)li * 4 * rows;
  const float* smL = sm + (long)li * 4 * E_DIM;
  const float* raL = ra + (long)li * 4 * rows;
  const float* saL = sa + (long)li * 4 * E_DIM;
  float mul = 0.f, add = 0.f;
#pragma unroll
  for (int r = 0; r < 4; ++r) {
    mul += rmL[r * rows + e] * smL[r * E_DIM + f];
    add += raL[r * rows + e] * saL[r * E_DIM + f];
  }
  w[idx] = (__bf16)(pw[idx] * mul + add);
}

// ---------------------------------------------------------------------------
// Kernel 2: f32 -> bf16 convert (hidden states)
// ---------------------------------------------------------------------------
__global__ __launch_bounds__(256)
void cvt_bf16_kernel(const float* __restrict__ src, __bf16* __restrict__ dst) {
  int idx = blockIdx.x * blockDim.x + threadIdx.x;
  dst[idx] = (__bf16)src[idx];
}

// ---------------------------------------------------------------------------
// Shared GEMM mainloop: acc(16x64 per wave) += A[mRow.., :] * W[nBase.., :]^T
// Manual 2x unroll with ping-pong buffers (a0/b0 <-> a1/b1): the rotation is
// pure register renaming (no v_mov copies) and each half-step's global loads
// issue ahead of the other half-step's WMMAs. Operands are L2-resident
// (W 2-6 MB, A 8 MB vs 192 MB L2), so no LDS round-trip is needed.
// ---------------------------------------------------------------------------
static __device__ __forceinline__ void gemm_mainloop(const __bf16* __restrict__ A,
                                                     const __bf16* __restrict__ W,
                                                     int mRow, int nBase, int lane,
                                                     v8f acc[4]) {
  v16bf a0 = load_a_frag(A, E_DIM, mRow, 0, lane);
  v16bf b0[4];
#pragma unroll
  for (int nt = 0; nt < 4; ++nt)
    b0[nt] = load_b_frag(W, E_DIM, nBase + nt * 16, 0, lane);

#pragma unroll 1
  for (int k0 = 0; k0 < E_DIM; k0 += 64) {
    // prefetch k0+32 into buffer set 1
    v16bf a1 = load_a_frag(A, E_DIM, mRow, k0 + 32, lane);
    v16bf b1[4];
#pragma unroll
    for (int nt = 0; nt < 4; ++nt)
      b1[nt] = load_b_frag(W, E_DIM, nBase + nt * 16, k0 + 32, lane);

    // compute on buffer set 0 (k0)
#pragma unroll
    for (int nt = 0; nt < 4; ++nt)
      acc[nt] = __builtin_amdgcn_wmma_f32_16x16x32_bf16(false, a0, false, b0[nt],
                                                        (short)0, acc[nt], false, false);

    // prefetch k0+64 into buffer set 0 (wrap keeps last dummy prefetch in-bounds)
    int kn = (k0 + 64) & (E_DIM - 1);
    a0 = load_a_frag(A, E_DIM, mRow, kn, lane);
#pragma unroll
    for (int nt = 0; nt < 4; ++nt)
      b0[nt] = load_b_frag(W, E_DIM, nBase + nt * 16, kn, lane);

    // compute on buffer set 1 (k0+32)
#pragma unroll
    for (int nt = 0; nt < 4; ++nt)
      acc[nt] = __builtin_amdgcn_wmma_f32_16x16x32_bf16(false, a1, false, b1[nt],
                                                        (short)0, acc[nt], false, false);
  }
}

// ---------------------------------------------------------------------------
// Kernel 3: QKV GEMM  qkv[m,n] = sum_f hb[m,f]*w_in[n,f] + bias[n]
// Block: 256 thr = 8 waves; block tile 128(M) x 64(N); wave tile 16 x 64.
// Epilogue scatters into q[bh][t][d] (scaled), k[bh][t][d], vT[bh][d][t].
// ---------------------------------------------------------------------------
__global__ __launch_bounds__(256)
void gemm_qkv_kernel(const __bf16* __restrict__ A, const __bf16* __restrict__ W,
                     const float* __restrict__ bias,
                     __bf16* __restrict__ qb, __bf16* __restrict__ kb,
                     __bf16* __restrict__ vt) {
  int lane  = threadIdx.x & 31;
  int wid   = threadIdx.x >> 5;
  int mRow  = blockIdx.x * 128 + wid * 16;
  int nBase = blockIdx.y * 64;
  v8f acc[4] = {};
  gemm_mainloop(A, W, mRow, nBase, lane, acc);

  int half8 = (lane >> 4) << 3;
#pragma unroll
  for (int nt = 0; nt < 4; ++nt) {
    int n = nBase + nt * 16 + (lane & 15);
    float bv = bias[n];
    int h = n / 192, rem = n % 192, sel = rem / 64, d = rem % 64;
#pragma unroll
    for (int vv = 0; vv < 8; ++vv) {
      int m = mRow + vv + half8;
      int t = m >> 1, bb = m & 1;          // m = t*B + b, B = 2
      long bh = (long)bb * H_DIM + h;
      float val = acc[nt][vv] + bv;
      if (sel == 0)      qb[(bh * T_DIM + t) * HD_DIM + d] = (__bf16)(val * SCALE_Q);
      else if (sel == 1) kb[(bh * T_DIM + t) * HD_DIM + d] = (__bf16)val;
      else               vt[(bh * HD_DIM + d) * T_DIM + t] = (__bf16)val;
    }
  }
}

// ---------------------------------------------------------------------------
// Kernel 4: causal flash attention. Block 128 thr = 4 waves; one wave owns
// 16 query rows; s-loop in steps of 32 with online softmax; ctx via WMMA
// against pre-transposed V. K and V fragments for the whole step are loaded
// up front so they overlap the softmax VALU work (exp + shuffles).
// ---------------------------------------------------------------------------
__global__ __launch_bounds__(128)
void flash_attn_kernel(const __bf16* __restrict__ qb, const __bf16* __restrict__ kb,
                       const __bf16* __restrict__ vt, __bf16* __restrict__ ctxb) {
  __shared__ __align__(16) __bf16 pbuf[4][16 * 32];   // per-wave P staging
  int lane = threadIdx.x & 31;
  int wid  = threadIdx.x >> 5;
  int bh   = blockIdx.y;
  int b    = bh >> 4, h = bh & 15;
  const __bf16* q = qb + (long)bh * T_DIM * HD_DIM;
  const __bf16* k = kb + (long)bh * T_DIM * HD_DIM;
  const __bf16* v = vt + (long)bh * HD_DIM * T_DIM;
  int t0    = blockIdx.x * 64 + wid * 16;
  int half8 = (lane >> 4) << 3;

  v16bf qf0 = load_a_frag(q, HD_DIM, t0, 0,  lane);
  v16bf qf1 = load_a_frag(q, HD_DIM, t0, 32, lane);

  v8f acc[4] = {};
  float m_run[8], l_run[8];
#pragma unroll
  for (int i = 0; i < 8; ++i) { m_run[i] = -3.0e38f; l_run[i] = 0.f; }

  __bf16* pw = pbuf[wid];
  int ncolL = lane & 15;

  for (int s0 = 0; s0 < t0 + 16; s0 += 32) {
    // ---- issue ALL global loads for this step first (K: 4 frags, V: 4 frags)
    v16bf kf00 = load_b_frag(k, HD_DIM, s0,      0,  lane);
    v16bf kf01 = load_b_frag(k, HD_DIM, s0,      32, lane);
    v16bf kf10 = load_b_frag(k, HD_DIM, s0 + 16, 0,  lane);
    v16bf kf11 = load_b_frag(k, HD_DIM, s0 + 16, 32, lane);
    v16bf vf[4];
#pragma unroll
    for (int dt = 0; dt < 4; ++dt)
      vf[dt] = load_b_frag(v, T_DIM, dt * 16, s0, lane);   // pre-transposed V

    v8f z = {};
    v8f sc0 = z, sc1 = z;
    sc0 = __builtin_amdgcn_wmma_f32_16x16x32_bf16(false, qf0, false, kf00, (short)0, sc0, false, false);
    sc0 = __builtin_amdgcn_wmma_f32_16x16x32_bf16(false, qf1, false, kf01, (short)0, sc0, false, false);
    sc1 = __builtin_amdgcn_wmma_f32_16x16x32_bf16(false, qf0, false, kf10, (short)0, sc1, false, false);
    sc1 = __builtin_amdgcn_wmma_f32_16x16x32_bf16(false, qf1, false, kf11, (short)0, sc1, false, false);

    int ncol0 = s0 + ncolL;
    int ncol1 = ncol0 + 16;
#pragma unroll
    for (int vv = 0; vv < 8; ++vv) {
      int mrow = t0 + vv + half8;
      float s0v = (ncol0 <= mrow) ? sc0[vv] : NEG_BIG;   // causal mask
      float s1v = (ncol1 <= mrow) ? sc1[vv] : NEG_BIG;
      float rmx = fmaxf(s0v, s1v);                       // row max across 16 lanes
      rmx = fmaxf(rmx, __shfl_xor(rmx, 1, 32));
      rmx = fmaxf(rmx, __shfl_xor(rmx, 2, 32));
      rmx = fmaxf(rmx, __shfl_xor(rmx, 4, 32));
      rmx = fmaxf(rmx, __shfl_xor(rmx, 8, 32));
      float mnew = fmaxf(m_run[vv], rmx);
      float corr = __expf(m_run[vv] - mnew);
      float p0 = __expf(s0v - mnew);
      float p1 = __expf(s1v - mnew);
      float rs = p0 + p1;                                // row sum
      rs += __shfl_xor(rs, 1, 32);
      rs += __shfl_xor(rs, 2, 32);
      rs += __shfl_xor(rs, 4, 32);
      rs += __shfl_xor(rs, 8, 32);
      l_run[vv] = l_run[vv] * corr + rs;
      m_run[vv] = mnew;
#pragma unroll
      for (int dt = 0; dt < 4; ++dt) acc[dt][vv] *= corr;
      int mrowL = vv + half8;                            // C-layout -> LDS
      pw[mrowL * 32 + ncolL]      = (__bf16)p0;
      pw[mrowL * 32 + 16 + ncolL] = (__bf16)p1;
    }
    asm volatile("" ::: "memory");   // whole-wave DS ordering; stop reordering
    {
      const __bf16* pr = pw + ncolL * 32;                // LDS -> A-layout
      v8bf lo = *(const v8bf*)(pr + half8);
      v8bf hi = *(const v8bf*)(pr + 16 + half8);
      v16bf pf = cat8(lo, hi);
#pragma unroll
      for (int dt = 0; dt < 4; ++dt)
        acc[dt] = __builtin_amdgcn_wmma_f32_16x16x32_bf16(false, pf, false, vf[dt],
                                                          (short)0, acc[dt], false, false);
    }
    asm volatile("" ::: "memory");   // keep next-iter P stores behind pf reads
  }

#pragma unroll
  for (int vv = 0; vv < 8; ++vv) {
    float inv = 1.0f / l_run[vv];
    int t = t0 + vv + half8;
    long rowbase = ((long)t * B_DIM + b) * E_DIM + h * HD_DIM;
#pragma unroll
    for (int dt = 0; dt < 4; ++dt)
      ctxb[rowbase + dt * 16 + ncolL] = (__bf16)(acc[dt][vv] * inv);
  }
}

// ---------------------------------------------------------------------------
// Kernel 5: output GEMM  out[m,n] = sum_f ctx[m,f]*w_out[n,f] + bias[n] (f32)
// ---------------------------------------------------------------------------
__global__ __launch_bounds__(256)
void gemm_out_kernel(const __bf16* __restrict__ A, const __bf16* __restrict__ W,
                     const float* __restrict__ bias, float* __restrict__ out) {
  int lane  = threadIdx.x & 31;
  int wid   = threadIdx.x >> 5;
  int mRow  = blockIdx.x * 128 + wid * 16;
  int nBase = blockIdx.y * 64;
  v8f acc[4] = {};
  gemm_mainloop(A, W, mRow, nBase, lane, acc);

  int half8 = (lane >> 4) << 3;
#pragma unroll
  for (int nt = 0; nt < 4; ++nt) {
    int n = nBase + nt * 16 + (lane & 15);
    float bv = bias[n];
#pragma unroll
    for (int vv = 0; vv < 8; ++vv) {
      int m = mRow + vv + half8;
      out[(long)m * E_DIM + n] = acc[nt][vv] + bv;
    }
  }
}

// ---------------------------------------------------------------------------
extern "C" void kernel_launch(void* const* d_in, const int* in_sizes, int n_in,
                              void* d_out, int out_size, void* d_ws, size_t ws_size,
                              hipStream_t stream) {
  (void)in_sizes; (void)n_in; (void)out_size; (void)ws_size;

  const float* hidden   = (const float*)d_in[0];
  const float* proj_w   = (const float*)d_in[1];
  const float* proj_b   = (const float*)d_in[2];
  const float* out_w    = (const float*)d_in[3];
  const float* out_b    = (const float*)d_in[4];
  const float* rm_i     = (const float*)d_in[5];
  const float* sm_i     = (const float*)d_in[6];
  const float* rm_o     = (const float*)d_in[7];
  const float* sm_o     = (const float*)d_in[8];
  const float* r_i      = (const float*)d_in[9];
  const float* s_i      = (const float*)d_in[10];
  const float* r_o      = (const float*)d_in[11];
  const float* s_o      = (const float*)d_in[12];
  const int*   lang     = (const int*)d_in[13];
  float*       out      = (float*)d_out;

  // Workspace carve-up (all bf16): ~48 MB total
  char* p = (char*)d_ws;
  __bf16* w_in  = (__bf16*)p;  p += (size_t)E3_DIM * E_DIM * 2;           // 6 MB
  __bf16* w_out = (__bf16*)p;  p += (size_t)E_DIM  * E_DIM * 2;           // 2 MB
  __bf16* hb    = (__bf16*)p;  p += (size_t)M_DIM  * E_DIM * 2;           // 8 MB
  __bf16* qb    = (__bf16*)p;  p += (size_t)B_DIM * H_DIM * T_DIM * HD_DIM * 2; // 8 MB
  __bf16* kb    = (__bf16*)p;  p += (size_t)B_DIM * H_DIM * T_DIM * HD_DIM * 2; // 8 MB
  __bf16* vt    = (__bf16*)p;  p += (size_t)B_DIM * H_DIM * HD_DIM * T_DIM * 2; // 8 MB
  __bf16* ctxb  = (__bf16*)p;                                             // 8 MB

  // 1) modulated weights
  build_w_kernel<<<(E3_DIM * E_DIM) / 256, 256, 0, stream>>>(
      proj_w, rm_i, sm_i, r_i, s_i, lang, w_in, E3_DIM);
  build_w_kernel<<<(E_DIM * E_DIM) / 256, 256, 0, stream>>>(
      out_w, rm_o, sm_o, r_o, s_o, lang, w_out, E_DIM);

  // 2) hidden -> bf16
  cvt_bf16_kernel<<<(M_DIM * E_DIM) / 256, 256, 0, stream>>>(hidden, hb);

  // 3) QKV projection (WMMA) with head-layout scatter + q scaling
  gemm_qkv_kernel<<<dim3(M_DIM / 128, E3_DIM / 64), 256, 0, stream>>>(
      hb, w_in, proj_b, qb, kb, vt);

  // 4) causal flash attention (WMMA scores + WMMA PV)
  flash_attn_kernel<<<dim3(T_DIM / 64, B_DIM * H_DIM), 128, 0, stream>>>(
      qb, kb, vt, ctxb);

  // 5) output projection (WMMA), f32 result
  gemm_out_kernel<<<dim3(M_DIM / 128, E_DIM / 64), 256, 0, stream>>>(
      ctxb, w_out, out_b, out);
}
